// Encoder_44083544326959
// MI455X (gfx1250) — matrix operand order repaired and checked by reading, not compiled, not run on hardware
//
#include <hip/hip_runtime.h>

#define N_NODESC 100000
#define DIMF 128
#define N_ETYPESC 3
#define N_EDGESC 250000
#define BN_EPS 1e-5f

typedef __attribute__((ext_vector_type(16))) _Float16 v16h;
typedef __attribute__((ext_vector_type(8)))  float    v8f;
typedef unsigned uint32x4_t __attribute__((ext_vector_type(4)));
typedef unsigned uint32x8_t __attribute__((ext_vector_type(8)));

union FragH {
  v16h v;
  _Float16 h[16];
  unsigned u[8];
  uint4 q[2];
};

// ---------------------------------------------------------------------------
// feat = n_emb[node_ind]
// ---------------------------------------------------------------------------
__global__ void gather_kernel(const float* __restrict__ n_emb,
                              const int* __restrict__ ind,
                              float* __restrict__ feat) {
  size_t idx = (size_t)blockIdx.x * blockDim.x + threadIdx.x;
  if (idx >= (size_t)N_NODESC * DIMF) return;
  int n = (int)(idx >> 7);
  int c = (int)(idx & 127);
  feat[idx] = n_emb[(size_t)ind[n] * DIMF + c];
}

// ---------------------------------------------------------------------------
// Pack a 128x128 fp32 weight (row-major, W[k][n]) into the exact B-fragment
// layout of V_WMMA_F32_16X16X32_F16 (ISA 7.12.2), as f16 pairs in u32:
//   index = ((kc*8 + nt)*32 + lane)*8 + j
//   lane<16 : col = nt*16+lane,    K = kc*32 + 2j + (j>=4 ? 8 : 0), +1
//   lane>=16: col = nt*16+lane-16, K = kc*32 + 8 + 2j + (j>=4 ? 8 : 0), +1
// Packed weights are TDM-copied to LDS by every GEMM block (L2-resident).
// ---------------------------------------------------------------------------
__global__ void pack_w_kernel(const float* __restrict__ W, unsigned* __restrict__ dst) {
  int i = blockIdx.x * blockDim.x + threadIdx.x;
  if (i >= 8192) return;
  int j    = i & 7;
  int lane = (i >> 3) & 31;
  int nt   = (i >> 8) & 7;
  int kc   = i >> 11;
  int n = nt * 16 + (lane & 15);
  int k = kc * 32 + (lane >> 4) * 8 + 2 * j + ((j >= 4) ? 8 : 0);
  union { _Float16 h[2]; unsigned u; } p;
  p.h[0] = (_Float16)W[(size_t)k * DIMF + n];
  p.h[1] = (_Float16)W[(size_t)(k + 1) * DIMF + n];
  dst[i] = p.u;
}

// ---------------------------------------------------------------------------
// Per-etype edge scatter: one wave per edge, float4 per lane (128 floats).
// MODE 0 (gcn):  m = feat[dst] - feat[src]*emb[et]
// MODE 1 (attn): m = feat[src]*emb[et]
// ---------------------------------------------------------------------------
template<int MODE>
__global__ void edge_kernel(const float* __restrict__ feat,
                            const float* __restrict__ e_emb,
                            const int* __restrict__ src,
                            const int* __restrict__ dst,
                            float* __restrict__ sums,
                            float* __restrict__ cnt) {
  int et  = blockIdx.y;
  int gid = blockIdx.x * blockDim.x + threadIdx.x;
  int e   = gid >> 5;
  if (e >= N_EDGESC) return;
  int lane = threadIdx.x & 31;
  int s = src[(size_t)et * N_EDGESC + e];
  int d = dst[(size_t)et * N_EDGESC + e];
  float4 vs = ((const float4*)(feat + (size_t)s * DIMF))[lane];
  float4 ve = ((const float4*)(e_emb + (size_t)et * DIMF))[lane];
  float4 m;
  if (MODE == 0) {
    float4 vd = ((const float4*)(feat + (size_t)d * DIMF))[lane];
    m.x = vd.x - vs.x * ve.x;
    m.y = vd.y - vs.y * ve.y;
    m.z = vd.z - vs.z * ve.z;
    m.w = vd.w - vs.w * ve.w;
  } else {
    m.x = vs.x * ve.x;
    m.y = vs.y * ve.y;
    m.z = vs.z * ve.z;
    m.w = vs.w * ve.w;
  }
  float* out = sums + ((size_t)et * N_NODESC + d) * DIMF + lane * 4;
  atomicAdd(out + 0, m.x);
  atomicAdd(out + 1, m.y);
  atomicAdd(out + 2, m.z);
  atomicAdd(out + 3, m.w);
  if (lane == 0) atomicAdd(&cnt[(size_t)et * N_NODESC + d], 1.0f);
}

// k[n][c] = sum_t sums[t][n][c] / max(cnt[t][n],1)
__global__ void norm_gcn_kernel(const float* __restrict__ sums,
                                const float* __restrict__ cnt,
                                float* __restrict__ k) {
  size_t idx = (size_t)blockIdx.x * blockDim.x + threadIdx.x;
  if (idx >= (size_t)N_NODESC * DIMF) return;
  int n = (int)(idx >> 7);
  const size_t st = (size_t)N_NODESC * DIMF;
  float r = 0.f;
#pragma unroll
  for (int t = 0; t < N_ETYPESC; ++t)
    r += sums[(size_t)t * st + idx] / fmaxf(cnt[(size_t)t * N_NODESC + n], 1.0f);
  k[idx] = r;
}

// in-place: key[t][n][c] = sums[t][n][c] / max(cnt[t][n],1)
__global__ void norm_attn_kernel(float* __restrict__ sums,
                                 const float* __restrict__ cnt) {
  size_t idx = (size_t)blockIdx.x * blockDim.x + threadIdx.x;
  if (idx >= (size_t)N_NODESC * DIMF) return;
  int n = (int)(idx >> 7);
  const size_t st = (size_t)N_NODESC * DIMF;
#pragma unroll
  for (int t = 0; t < N_ETYPESC; ++t) {
    float inv = 1.0f / fmaxf(cnt[(size_t)t * N_NODESC + n], 1.0f);
    sums[(size_t)t * st + idx] *= inv;
  }
}

// Column-wise partial sums/sumsq for BatchNorm (128 cols, coalesced rows).
__global__ void bn_partial_kernel(const float* __restrict__ h,
                                  float* __restrict__ cs,
                                  float* __restrict__ cq) {
  int c  = threadIdx.x;
  int r0 = blockIdx.x * 512;
  int r1 = (r0 + 512 < N_NODESC) ? r0 + 512 : N_NODESC;
  float s = 0.f, q = 0.f;
  for (int r = r0; r < r1; ++r) {
    float v = h[(size_t)r * DIMF + c];
    s += v;
    q += v * v;
  }
  atomicAdd(&cs[c], s);
  atomicAdd(&cq[c], q);
}

// scale/shift so BN+affine folds into a single fma in the next GEMM A-load.
__global__ void bn_finalize_kernel(const float* __restrict__ cs,
                                   const float* __restrict__ cq,
                                   const float* __restrict__ gamma,
                                   const float* __restrict__ beta,
                                   float* __restrict__ scale,
                                   float* __restrict__ sbias) {
  int c = threadIdx.x;
  const float invN = 1.0f / (float)N_NODESC;
  float mean = cs[c] * invN;
  float var  = cq[c] * invN - mean * mean;   // biased var, matches jnp.var
  float s = gamma[c] * rsqrtf(var + BN_EPS);
  scale[c] = s;
  sbias[c] = beta[c] - mean * s;
}

// softmax over the 3 etype scores per (n,h), weighted sum of v.
__global__ void softmax_o_kernel(const float* __restrict__ s3,
                                 const float* __restrict__ v3,
                                 float* __restrict__ o) {
  size_t idx = (size_t)blockIdx.x * blockDim.x + threadIdx.x;
  if (idx >= (size_t)N_NODESC * DIMF) return;
  const size_t st = (size_t)N_NODESC * DIMF;
  float a0 = s3[idx], a1 = s3[idx + st], a2 = s3[idx + 2 * st];
  float m  = fmaxf(a0, fmaxf(a1, a2));
  float e0 = __expf(a0 - m), e1 = __expf(a1 - m), e2 = __expf(a2 - m);
  float inv = 1.0f / (e0 + e1 + e2);
  o[idx] = (e0 * v3[idx] + e1 * v3[idx + st] + e2 * v3[idx + 2 * st]) * inv;
}

// ---------------------------------------------------------------------------
// WMMA GEMM: C[nRows x 128] = op(A) @ W  (+ epilogue)
// 8 waves/block, each wave a 16x128 strip: 8 v8f accumulators, K = 4 x K32
// -> 32 v_wmma_f32_16x16x32_f16 per wave. The packed 32KB B matrix is staged
// into LDS once per block by the Tensor Data Mover (tensor_load_to_lds),
// so every B fragment is a low-latency ds_load_b128 pair instead of an L2
// round trip.
// AMODE: 0 = plain, 1 = relu(a*scale[c]+shift[c]) (fused BN+ReLU), 2 = A - A1
// EPI:   0 = none, 1 = +bias[col], 2 = +bias[col] + res[row][col]
// ---------------------------------------------------------------------------
template<int AMODE, int EPI>
__global__ __launch_bounds__(256) void gemm128_wmma(
    const float* __restrict__ A, const float* __restrict__ A1,
    const unsigned* __restrict__ Bp, float* C,
    const float* res, const float* __restrict__ bias,
    const float* __restrict__ sc, const float* __restrict__ sb, int nRows) {
  __shared__ unsigned sB[8192];  // 32KB packed f16 weight in B-fragment layout

  // --- TDM stage: one wave DMAs the 32KB weight tile global->LDS ----------
  if (threadIdx.x == 0) {
    unsigned long long ga = (unsigned long long)Bp;      // 57-bit byte address
    unsigned lds_off = (unsigned)(size_t)(void*)sB;      // low 32 bits = LDS offset
    // D# group 0: count=1 | lds_addr | global_addr[56:0] | type=2 ("image")
    uint32x4_t g0;
    g0[0] = 1u;
    g0[1] = lds_off;
    g0[2] = (unsigned)ga;
    g0[3] = (unsigned)((ga >> 32) & 0x1FFFFFFu) | (2u << 30);
    // D# group 1: data_size=2 (4B); tensor_dim0=8192 (bits 79:48);
    // tensor_dim1=1 (bits 111:80); tile_dim0=8192 (bits 127:112);
    // tile_dim1=0 (1-D tile); tensor_dim0_stride=8192 (bits 207:160)
    uint32x8_t g1;
    g1[0] = 2u << 16;
    g1[1] = (8192u & 0xFFFFu) << 16;
    g1[2] = (8192u >> 16) | (1u << 16);
    g1[3] = 8192u << 16;
    g1[4] = 0u;
    g1[5] = 8192u;
    g1[6] = 0u;
    g1[7] = 0u;
    asm volatile("tensor_load_to_lds %0, %1" :: "s"(g0), "s"(g1) : "memory");
    __builtin_amdgcn_s_wait_tensorcnt(0);
  }
  __syncthreads();

  const int lane   = threadIdx.x & 31;
  const int wave   = threadIdx.x >> 5;
  const int lane16 = lane & 15;
  const int khalf  = lane >> 4;
  const int rowBase = blockIdx.x * 128 + wave * 16;

  int r  = rowBase + lane16;
  int rc = (r < nRows) ? r : (nRows - 1);   // clamp: keep EXEC all-ones for WMMA
  const float* arow  = A + (size_t)rc * DIMF;
  const float* a1row = (AMODE == 2) ? (A1 + (size_t)rc * DIMF) : nullptr;

  v8f acc[8] = {};

#pragma unroll
  for (int kc = 0; kc < 4; ++kc) {
    const int k0 = kc * 32 + khalf * 8;
    if (kc < 3) __builtin_prefetch(arow + k0 + 32, 0, 3);  // global_prefetch_b8

    // A fragment per ISA 16-bit 16x32 layout:
    //   h[0..7]  = K k0..k0+7, h[8..15] = K k0+16..k0+23
    FragH af;
#pragma unroll
    for (int j = 0; j < 8; ++j) {
      float x0 = arow[k0 + j];
      float x1 = arow[k0 + 16 + j];
      if (AMODE == 1) {
        x0 = fmaxf(fmaf(x0, sc[k0 + j],      sb[k0 + j]),      0.0f);
        x1 = fmaxf(fmaf(x1, sc[k0 + 16 + j], sb[k0 + 16 + j]), 0.0f);
      } else if (AMODE == 2) {
        x0 -= a1row[k0 + j];
        x1 -= a1row[k0 + 16 + j];
      }
      af.h[j]     = (_Float16)x0;
      af.h[8 + j] = (_Float16)x1;
    }

#pragma unroll
    for (int nt = 0; nt < 8; ++nt) {
      FragH bf;
      const uint4* bq = (const uint4*)(sB + (((kc * 8 + nt) * 32 + lane) << 3));
      bf.q[0] = bq[0];   // ds_load_b128
      bf.q[1] = bq[1];   // ds_load_b128
      acc[nt] = __builtin_amdgcn_wmma_f32_16x16x32_f16(
          false, af.v, false, bf.v, (short)0, acc[nt], false, false);
    }
  }

  // C/D layout: VGPR rr -> row rr + 8*khalf, col = lane16 (per 16x16 tile)
#pragma unroll
  for (int nt = 0; nt < 8; ++nt) {
    int col = nt * 16 + lane16;
    float bcol = (EPI >= 1) ? bias[col] : 0.0f;
#pragma unroll
    for (int rr = 0; rr < 8; ++rr) {
      int row = rowBase + khalf * 8 + rr;
      if (row < nRows) {
        float v = acc[nt][rr] + bcol;
        if (EPI == 2) v += res[(size_t)row * DIMF + col];
        C[(size_t)row * DIMF + col] = v;
      }
    }
  }
}

// ---------------------------------------------------------------------------
extern "C" void kernel_launch(void* const* d_in, const int* in_sizes, int n_in,
                              void* d_out, int out_size, void* d_ws, size_t ws_size,
                              hipStream_t stream) {
  (void)in_sizes; (void)n_in; (void)out_size; (void)ws_size;

  const float* n_emb   = (const float*)d_in[0];
  const float* e_emb   = (const float*)d_in[1];
  const int*   node_ind = (const int*)d_in[26];
  const int*   edge_src = (const int*)d_in[27];
  const int*   edge_dst = (const int*)d_in[28];

  const size_t NF = (size_t)N_NODESC * DIMF;
  float* ws    = (float*)d_ws;
  float* FEAT  = ws;             // N x 128
  float* BUFA  = FEAT + NF;      // k (gcn aggregate)
  float* BUFB  = BUFA + NF;      // h (gcn hidden)
  float* BUFQ  = BUFB + NF;      // q / o (attn)
  float* SUMS  = BUFQ + NF;      // [3][N][128]: seg sums -> key -> scores
  float* KK    = SUMS + 3 * NF;  // [3][N][128]
  float* VV    = KK   + 3 * NF;  // [3][N][128]
  float* CNT   = VV   + 3 * NF;  // [3][N]
  float* COLS  = CNT  + 3 * N_NODESC;
  float* COLQ  = COLS + DIMF;
  float* SCALE = COLQ + DIMF;
  float* SBIAS = SCALE + DIMF;
  unsigned* PW = (unsigned*)(SBIAS + DIMF);  // 14 packed weights x 8192 u32

  // Pack all 14 weight matrices into WMMA B-fragment layout (f16).
  const float* wsrc[14] = {
    (const float*)d_in[2],  (const float*)d_in[5],                      // m1_w1, m1_w2
    (const float*)d_in[7],  (const float*)d_in[8],  (const float*)d_in[9],
    (const float*)d_in[10], (const float*)d_in[12],                     // a1 wq wk wv wa wp
    (const float*)d_in[14], (const float*)d_in[17],                     // m2_w1, m2_w2
    (const float*)d_in[19], (const float*)d_in[20], (const float*)d_in[21],
    (const float*)d_in[22], (const float*)d_in[24],                     // a2 wq wk wv wa wp
  };
  for (int i = 0; i < 14; ++i)
    pack_w_kernel<<<32, 256, 0, stream>>>(wsrc[i], PW + (size_t)i * 8192);

  const dim3 elemGrid((unsigned)((NF + 255) / 256));
  const dim3 edgeGrid((N_EDGESC * 32 + 255) / 256, N_ETYPESC);
  const int  gemmGrid = (N_NODESC + 127) / 128;

  gather_kernel<<<elemGrid, 256, 0, stream>>>(n_emb, node_ind, FEAT);

  auto gcn = [&](const unsigned* pw1, const unsigned* pw2,
                 const float* gamma, const float* beta, const float* b2) {
    hipMemsetAsync(SUMS, 0, 3 * NF * sizeof(float), stream);
    hipMemsetAsync(CNT,  0, 3 * N_NODESC * sizeof(float), stream);
    edge_kernel<0><<<edgeGrid, 256, 0, stream>>>(FEAT, e_emb, edge_src, edge_dst, SUMS, CNT);
    norm_gcn_kernel<<<elemGrid, 256, 0, stream>>>(SUMS, CNT, BUFA);
    gemm128_wmma<0, 0><<<gemmGrid, 256, 0, stream>>>(
        BUFA, nullptr, pw1, BUFB, nullptr, nullptr, nullptr, nullptr, N_NODESC);
    hipMemsetAsync(COLS, 0, 2 * DIMF * sizeof(float), stream);
    bn_partial_kernel<<<(N_NODESC + 511) / 512, 128, 0, stream>>>(BUFB, COLS, COLQ);
    bn_finalize_kernel<<<1, 128, 0, stream>>>(COLS, COLQ, gamma, beta, SCALE, SBIAS);
    // feat += relu(BN(h)) @ w2 + b2, BN+ReLU fused into A-load
    gemm128_wmma<1, 2><<<gemmGrid, 256, 0, stream>>>(
        BUFB, nullptr, pw2, FEAT, FEAT, b2, SCALE, SBIAS, N_NODESC);
  };

  auto attn = [&](const unsigned* pwq, const unsigned* pwk, const unsigned* pwv,
                  const unsigned* pwa, const float* ba,
                  const unsigned* pwp, const float* bp) {
    hipMemsetAsync(SUMS, 0, 3 * NF * sizeof(float), stream);
    hipMemsetAsync(CNT,  0, 3 * N_NODESC * sizeof(float), stream);
    edge_kernel<1><<<edgeGrid, 256, 0, stream>>>(FEAT, e_emb, edge_src, edge_dst, SUMS, CNT);
    norm_attn_kernel<<<elemGrid, 256, 0, stream>>>(SUMS, CNT);          // SUMS -> key
    gemm128_wmma<0, 0><<<gemmGrid, 256, 0, stream>>>(
        FEAT, nullptr, pwq, BUFQ, nullptr, nullptr, nullptr, nullptr, N_NODESC);  // q
    for (int t = 0; t < N_ETYPESC; ++t) {
      gemm128_wmma<0, 0><<<gemmGrid, 256, 0, stream>>>(
          SUMS + t * NF, nullptr, pwk, KK + t * NF, nullptr, nullptr, nullptr, nullptr, N_NODESC);
      gemm128_wmma<0, 0><<<gemmGrid, 256, 0, stream>>>(
          SUMS + t * NF, nullptr, pwv, VV + t * NF, nullptr, nullptr, nullptr, nullptr, N_NODESC);
    }
    // scores[t] = (q - k[t]) @ wa + ba   (diff fused into A-load; overwrites key)
    for (int t = 0; t < N_ETYPESC; ++t)
      gemm128_wmma<2, 1><<<gemmGrid, 256, 0, stream>>>(
          BUFQ, KK + t * NF, pwa, SUMS + t * NF, nullptr, ba, nullptr, nullptr, N_NODESC);
    softmax_o_kernel<<<elemGrid, 256, 0, stream>>>(SUMS, VV, BUFQ);     // o (overwrites q)
    gemm128_wmma<0, 2><<<gemmGrid, 256, 0, stream>>>(
        BUFQ, nullptr, pwp, FEAT, FEAT, bp, nullptr, nullptr, N_NODESC);
  };

  gcn (PW + 0 * 8192, PW + 1 * 8192, (const float*)d_in[3],  (const float*)d_in[4],  (const float*)d_in[6]);
  attn(PW + 2 * 8192, PW + 3 * 8192, PW + 4 * 8192, PW + 5 * 8192,
       (const float*)d_in[11], PW + 6 * 8192, (const float*)d_in[13]);
  gcn (PW + 7 * 8192, PW + 8 * 8192, (const float*)d_in[15], (const float*)d_in[16], (const float*)d_in[18]);
  attn(PW + 9 * 8192, PW + 10 * 8192, PW + 11 * 8192, PW + 12 * 8192,
       (const float*)d_in[23], PW + 13 * 8192, (const float*)d_in[25]);

  // Output: (feat, e_emb) concatenated flat.
  hipMemcpyAsync(d_out, FEAT, NF * sizeof(float), hipMemcpyDeviceToDevice, stream);
  hipMemcpyAsync((float*)d_out + NF, e_emb, (size_t)N_ETYPESC * DIMF * sizeof(float),
                 hipMemcpyDeviceToDevice, stream);
}
